// ActorNetwork_26345329393717
// MI455X (gfx1250) — matrix operand order repaired and checked by reading, not compiled, not run on hardware
//
#include <hip/hip_runtime.h>
#include <math.h>

// ---------------------------------------------------------------------------
// CDNA5 WMMA types (wave32).
// ---------------------------------------------------------------------------
typedef __bf16 bf16x16 __attribute__((ext_vector_type(16)));
typedef __bf16 bf16x8  __attribute__((ext_vector_type(8)));
typedef float  f32x8   __attribute__((ext_vector_type(8)));

#define F_IN  128
#define HID   64
#define HEADS 4
#define F1    (HEADS * HID)   // 256

// ---------------------------------------------------------------------------
// Helpers
// ---------------------------------------------------------------------------
__device__ __forceinline__ void atomicMaxF(float* addr, float val) {
  // Two-path punning trick; memory initialized to -INF (0xFF800000 works
  // for both the signed-int max path and the unsigned min path).
  if (val >= 0.0f) {
    atomicMax((int*)addr, __float_as_int(val));
  } else {
    atomicMin((unsigned int*)addr, __float_as_uint(val));
  }
}

__device__ __forceinline__ float lrelu02(float v) {
  return v > 0.0f ? v : 0.2f * v;
}

__global__ void fill_f32(float* __restrict__ p, float v, size_t n) {
  size_t i = (size_t)blockIdx.x * blockDim.x + threadIdx.x;
  if (i < n) p[i] = v;
}

// f32 -> bf16 elementwise convert (prep pass; L2 resident).
__global__ void cvt_bf16(const float* __restrict__ src, __bf16* __restrict__ dst,
                         size_t n) {
  size_t i = (size_t)blockIdx.x * blockDim.x + threadIdx.x;
  if (i < n) dst[i] = (__bf16)src[i];
}

// W[K,Nout] f32 -> Wt[Nout,K] bf16 (tiny; makes B fragments contiguous).
__global__ void wtrans_bf16(const float* __restrict__ W, __bf16* __restrict__ Wt,
                            int K, int Nout) {
  int t = blockIdx.x * blockDim.x + threadIdx.x;
  if (t >= K * Nout) return;
  int n = t / K, k = t - n * K;
  Wt[t] = (__bf16)W[(size_t)k * Nout + n];
}

// h1r[N,256] = bf16(relu(h1agg + b1))  (layer-2 GEMM A operand prep).
__global__ void relu_bias_bf16(const float* __restrict__ h,
                               const float* __restrict__ b1,
                               __bf16* __restrict__ out, size_t total) {
  size_t i = (size_t)blockIdx.x * blockDim.x + threadIdx.x;
  if (i >= total) return;
  const int c = (int)(i & (F1 - 1));
  float v = h[i] + b1[c];
  out[i] = (__bf16)(v > 0.0f ? v : 0.0f);
}

// ---------------------------------------------------------------------------
// Generic bf16 WMMA GEMM: C[N,NOUT] = A[N,K] @ Bt[NOUT,K]^T
// One wave computes a 16x16 tile. Per ISA 7.12.2:
//   A fragment (lane, half h): K runs [kt+8h, +8) and [kt+16+8h, +8)  -> 2 x b128
//   B fragment (lane, half h): K runs [kt+8h, +8) and [kt+16+8h, +8)  -> 2 x b128
// grid(ceil(N/16), NOUT/64), block(128): wave w -> col tile 4*by + w.
// ---------------------------------------------------------------------------
template <int K, int NOUT>
__global__ void gemm_wmma(const __bf16* __restrict__ A,
                          const __bf16* __restrict__ Bt,
                          float* __restrict__ C, int Nn) {
  const int wave = threadIdx.x >> 5;
  const int lane = threadIdx.x & 31;
  const int half = lane >> 4;
  const int ml   = lane & 15;
  const int m0   = blockIdx.x * 16;
  const int n0   = (blockIdx.y * 4 + wave) * 16;

  int rowc = m0 + ml;
  if (rowc >= Nn) rowc = Nn - 1;  // clamp loads; stores handled below

  const __bf16* __restrict__ arow = A + (size_t)rowc * K;
  const __bf16* __restrict__ brow = Bt + (size_t)(n0 + ml) * K;

  f32x8 acc = {};
#pragma unroll
  for (int kt = 0; kt < K; kt += 32) {
    const bf16x8 a0 = *(const bf16x8*)(arow + kt + 8 * half);
    const bf16x8 a1 = *(const bf16x8*)(arow + kt + 16 + 8 * half);
    const bf16x8 b0 = *(const bf16x8*)(brow + kt + 8 * half);
    const bf16x8 b1 = *(const bf16x8*)(brow + kt + 16 + 8 * half);
    const bf16x16 a = __builtin_shufflevector(a0, a1, 0, 1, 2, 3, 4, 5, 6, 7,
                                              8, 9, 10, 11, 12, 13, 14, 15);
    const bf16x16 b = __builtin_shufflevector(b0, b1, 0, 1, 2, 3, 4, 5, 6, 7,
                                              8, 9, 10, 11, 12, 13, 14, 15);
    acc = __builtin_amdgcn_wmma_f32_16x16x32_bf16(false, a, false, b,
                                                  (short)0, acc, false, false);
  }

  // C fragment: VGPR r <-> row m0 + r + 8*half, col n0 + ml. One base pointer,
  // compile-time byte offsets; wave-uniform fast path avoids EXEC churn.
  float* __restrict__ crow = C + (size_t)(m0 + 8 * half) * NOUT + n0 + ml;
  if (m0 + 16 <= Nn) {
#pragma unroll
    for (int r = 0; r < 8; ++r) crow[(size_t)r * NOUT] = acc[r];
  } else {
#pragma unroll
    for (int r = 0; r < 8; ++r) {
      if (m0 + r + 8 * half < Nn) crow[(size_t)r * NOUT] = acc[r];
    }
  }
}

// ---------------------------------------------------------------------------
// Attention logits, layer 1: a_s[n,h] = <h1[n,h,:], att_src1[h,:]>, same a_d.
// ---------------------------------------------------------------------------
__global__ void att_logits1(const float* __restrict__ h1,
                            const float* __restrict__ att_s,
                            const float* __restrict__ att_d,
                            float* __restrict__ as1, float* __restrict__ ad1,
                            int Nn) {
  const int t = blockIdx.x * blockDim.x + threadIdx.x;
  if (t >= Nn * HEADS) return;
  const int n = t >> 2, hd = t & 3;
  const float* hp = h1 + (size_t)n * F1 + hd * HID;
  const float* sp = att_s + hd * HID;
  const float* dp = att_d + hd * HID;
  float s = 0.0f, d = 0.0f;
#pragma unroll 8
  for (int c = 0; c < HID; ++c) { const float v = hp[c]; s += v * sp[c]; d += v * dp[c]; }
  as1[t] = s; ad1[t] = d;
}

// Attention logits, layer 2 (single head over h2[N,64]).
__global__ void att_logits2(const float* __restrict__ h2,
                            const float* __restrict__ att_s,
                            const float* __restrict__ att_d,
                            float* __restrict__ as2, float* __restrict__ ad2,
                            int Nn) {
  const int n = blockIdx.x * blockDim.x + threadIdx.x;
  if (n >= Nn) return;
  const float* hp = h2 + (size_t)n * HID;
  float s = 0.0f, d = 0.0f;
#pragma unroll 8
  for (int c = 0; c < HID; ++c) { const float v = hp[c]; s += v * att_s[c]; d += v * att_d[c]; }
  as2[n] = s; ad2[n] = d;
}

// ---------------------------------------------------------------------------
// Edge passes (edges + implicit self loops at index >= E).
// ---------------------------------------------------------------------------
__global__ void edge_maxk(const int* __restrict__ ei, int E, int Nn, int H,
                          const float* __restrict__ as, const float* __restrict__ ad,
                          float* __restrict__ m) {
  const int i = blockIdx.x * blockDim.x + threadIdx.x;
  const int EE = E + Nn;
  if (i >= EE) return;
  int s, d;
  if (i < E) { s = ei[i]; d = ei[E + i]; } else { s = d = i - E; }
  for (int hd = 0; hd < H; ++hd) {
    const float v = lrelu02(as[(size_t)s * H + hd] + ad[(size_t)d * H + hd]);
    atomicMaxF(&m[(size_t)d * H + hd], v);
  }
}

__global__ void edge_expsum(const int* __restrict__ ei, int E, int Nn, int H,
                            const float* __restrict__ as, const float* __restrict__ ad,
                            const float* __restrict__ m, float* __restrict__ dnm) {
  const int i = blockIdx.x * blockDim.x + threadIdx.x;
  const int EE = E + Nn;
  if (i >= EE) return;
  int s, d;
  if (i < E) { s = ei[i]; d = ei[E + i]; } else { s = d = i - E; }
  for (int hd = 0; hd < H; ++hd) {
    const float v = lrelu02(as[(size_t)s * H + hd] + ad[(size_t)d * H + hd]);
    atomicAdd(&dnm[(size_t)d * H + hd], expf(v - m[(size_t)d * H + hd]));
  }
}

// One block per edge; blockDim = H*64 features. Coefs broadcast via LDS.
__global__ void edge_aggr(const int* __restrict__ ei, int E, int Nn, int H,
                          const float* __restrict__ as, const float* __restrict__ ad,
                          const float* __restrict__ m, const float* __restrict__ dnm,
                          const float* __restrict__ hsrc, float* __restrict__ hagg) {
  const int e = blockIdx.x;
  int s, d;
  if (e < E) { s = ei[e]; d = ei[E + e]; } else { s = d = e - E; }
  const int t = threadIdx.x;
  const int F = H * HID;
  __shared__ float coef[HEADS];
  if (t < H) {
    const float v = lrelu02(as[(size_t)s * H + t] + ad[(size_t)d * H + t]);
    coef[t] = expf(v - m[(size_t)d * H + t]) / dnm[(size_t)d * H + t];
  }
  __syncthreads();
  const float c = coef[t >> 6];
  atomicAdd(&hagg[(size_t)d * F + t], c * hsrc[(size_t)s * F + t]);
}

// ---------------------------------------------------------------------------
// node_scores[n] = tanh(<h2agg[n]+b2, action_w> + action_b)
// ---------------------------------------------------------------------------
__global__ void node_scores_k(const float* __restrict__ h2agg,
                              const float* __restrict__ b2,
                              const float* __restrict__ aw,
                              const float* __restrict__ ab,
                              float* __restrict__ ns, int Nn) {
  const int n = blockIdx.x * blockDim.x + threadIdx.x;
  if (n >= Nn) return;
  const float* hp = h2agg + (size_t)n * HID;
  float s = 0.0f;
#pragma unroll 8
  for (int c = 0; c < HID; ++c) s += (hp[c] + b2[c]) * aw[c];
  ns[n] = tanhf(s + ab[0]);
}

// ---------------------------------------------------------------------------
// Per-graph context (masked gather-mean over cur_nodes) -> stop score.
// grid(G), block(64).
// ---------------------------------------------------------------------------
__global__ void ctx_stop_k(const float* __restrict__ h2agg,
                           const float* __restrict__ b2,
                           const int* __restrict__ cur_nodes,
                           const int* __restrict__ cur_counts,
                           const float* __restrict__ stop_w,
                           const float* __restrict__ stop_b,
                           float* __restrict__ stopsc, int Kk) {
  const int g = blockIdx.x;
  const int c = threadIdx.x;  // 0..63
  const int cnt = cur_counts[g];
  float acc = 0.0f;
  for (int k = 0; k < Kk; ++k) {
    if (k < cnt) {
      const int n = cur_nodes[g * Kk + k];
      acc += h2agg[(size_t)n * HID + c] + b2[c];
    }
  }
  const float denom = (float)(cnt > 1 ? cnt : 1);
  __shared__ float red[HID];
  red[c] = (acc / denom) * stop_w[c];
  __syncthreads();
  for (int s = HID / 2; s > 0; s >>= 1) {
    if (c < s) red[c] += red[c + s];
    __syncthreads();
  }
  if (c == 0) stopsc[g] = tanhf(red[0] + stop_b[0]);
}

// ---------------------------------------------------------------------------
// Final masked softmax over [G, N+1]; cur_counts==0 -> one-hot at N.
// grid(G), block(256).
// ---------------------------------------------------------------------------
__global__ void softmax_out_k(const float* __restrict__ ns,
                              const unsigned char* __restrict__ masks,
                              const float* __restrict__ stopsc,
                              const int* __restrict__ cur_counts,
                              float* __restrict__ out, int Nn) {
  const int g = blockIdx.x;
  const int t = threadIdx.x;
  float* og = out + (size_t)g * (Nn + 1);
  __shared__ float red[256];

  if (cur_counts[g] == 0) {
    for (int j = t; j < Nn; j += 256) og[j] = 0.0f;
    if (t == 0) og[Nn] = 1.0f;
    return;
  }
  const unsigned char* mg = masks + (size_t)g * Nn;
  const float ss = stopsc[g];

  // pass 1: max
  float mx = ss;
  for (int j = t; j < Nn; j += 256)
    if (mg[j]) mx = fmaxf(mx, ns[j]);
  red[t] = mx; __syncthreads();
  for (int s = 128; s > 0; s >>= 1) { if (t < s) red[t] = fmaxf(red[t], red[t + s]); __syncthreads(); }
  mx = red[0]; __syncthreads();

  // pass 2: sum of exp
  float sm = (t == 0) ? expf(ss - mx) : 0.0f;
  for (int j = t; j < Nn; j += 256)
    sm += mg[j] ? expf(ns[j] - mx) : 0.0f;  // masked entries exp(-1e9-mx)==0
  red[t] = sm; __syncthreads();
  for (int s = 128; s > 0; s >>= 1) { if (t < s) red[t] += red[t + s]; __syncthreads(); }
  const float inv = 1.0f / red[0];

  // pass 3: write
  for (int j = t; j < Nn; j += 256)
    og[j] = mg[j] ? expf(ns[j] - mx) * inv : 0.0f;
  if (t == 0) og[Nn] = expf(ss - mx) * inv;
}

// ---------------------------------------------------------------------------
// Host launcher
// ---------------------------------------------------------------------------
extern "C" void kernel_launch(void* const* d_in, const int* in_sizes, int n_in,
                              void* d_out, int out_size, void* d_ws, size_t ws_size,
                              hipStream_t stream) {
  const float* x          = (const float*)d_in[0];
  const int*   ei         = (const int*)d_in[1];
  const int*   cur_nodes  = (const int*)d_in[2];
  const int*   cur_counts = (const int*)d_in[3];
  const unsigned char* masks = (const unsigned char*)d_in[4];
  const float* W1        = (const float*)d_in[5];
  const float* att_src1  = (const float*)d_in[6];
  const float* att_dst1  = (const float*)d_in[7];
  const float* b1        = (const float*)d_in[8];
  const float* W2        = (const float*)d_in[9];
  const float* att_src2  = (const float*)d_in[10];
  const float* att_dst2  = (const float*)d_in[11];
  const float* b2        = (const float*)d_in[12];
  const float* action_w  = (const float*)d_in[13];
  const float* action_b  = (const float*)d_in[14];
  const float* stop_w    = (const float*)d_in[15];
  const float* stop_b    = (const float*)d_in[16];

  const int Nn = in_sizes[0] / F_IN;
  const int E  = in_sizes[1] / 2;
  const int G  = in_sizes[3];
  const int Kk = in_sizes[2] / G;
  const int EE = E + Nn;

  // Workspace layout. Float regions first, then 16B-aligned bf16 regions.
  // h1 region is reused for h2/h2agg after layer-1 aggregation consumed h1.
  float* f = (float*)d_ws;
  size_t o = 0;
  float* h1    = f + o; o += (size_t)Nn * F1;      // [N,256] pre-agg features
  float* h1agg = f + o; o += (size_t)Nn * F1;      // [N,256] aggregated
  float* as1   = f + o; o += (size_t)Nn * HEADS;
  float* ad1   = f + o; o += (size_t)Nn * HEADS;
  float* m1    = f + o; o += (size_t)Nn * HEADS;
  float* d1    = f + o; o += (size_t)Nn * HEADS;
  float* as2   = f + o; o += (size_t)Nn;
  float* ad2   = f + o; o += (size_t)Nn;
  float* m2    = f + o; o += (size_t)Nn;
  float* d2    = f + o; o += (size_t)Nn;
  float* ns    = f + o; o += (size_t)Nn;
  float* stopsc = f + o; o += (size_t)G;
  o = (o + 31) & ~(size_t)31;                      // 128B align for bf16 blobs
  __bf16* xb  = (__bf16*)(f + o); o += (size_t)Nn * F_IN / 2;  // [N,128] bf16
  __bf16* w1t = (__bf16*)(f + o); o += (size_t)F1 * F_IN / 2;  // [256,128] bf16
  __bf16* w2t = (__bf16*)(f + o); o += (size_t)HID * F1 / 2;   // [64,256] bf16
  __bf16* h1r = (__bf16*)(f + o); o += (size_t)Nn * F1 / 2;    // [N,256] bf16
  float* h2    = h1;                                // [N,64]  aliases h1[0:64N)
  float* h2agg = h1 + (size_t)Nn * HID;             // [N,64]  aliases h1[64N:128N)

  const int mt = (Nn + 15) / 16;
  auto fill = [&](float* p, float v, size_t n) {
    fill_f32<<<dim3((unsigned)((n + 255) / 256)), dim3(256), 0, stream>>>(p, v, n);
  };

  // ----- operand prep (bf16, contiguous-fragment layouts) -----
  cvt_bf16<<<dim3((unsigned)(((size_t)Nn * F_IN + 255) / 256)), dim3(256), 0, stream>>>(
      x, xb, (size_t)Nn * F_IN);
  wtrans_bf16<<<dim3((F_IN * F1 + 255) / 256), dim3(256), 0, stream>>>(W1, w1t, F_IN, F1);
  wtrans_bf16<<<dim3((F1 * HID + 255) / 256), dim3(256), 0, stream>>>(W2, w2t, F1, HID);

  // ----- layer 1 -----
  fill(h1agg, 0.0f, (size_t)Nn * F1);
  fill(m1, -INFINITY, (size_t)Nn * HEADS);
  fill(d1, 0.0f, (size_t)Nn * HEADS);

  gemm_wmma<F_IN, F1><<<dim3(mt, 4), dim3(128), 0, stream>>>(xb, w1t, h1, Nn);
  att_logits1<<<dim3((Nn * HEADS + 255) / 256), dim3(256), 0, stream>>>(
      h1, att_src1, att_dst1, as1, ad1, Nn);
  edge_maxk<<<dim3((EE + 255) / 256), dim3(256), 0, stream>>>(ei, E, Nn, HEADS, as1, ad1, m1);
  edge_expsum<<<dim3((EE + 255) / 256), dim3(256), 0, stream>>>(ei, E, Nn, HEADS, as1, ad1, m1, d1);
  edge_aggr<<<dim3(EE), dim3(F1), 0, stream>>>(ei, E, Nn, HEADS, as1, ad1, m1, d1, h1, h1agg);

  // ----- layer 2 (h1 region now free -> h2/h2agg alias it) -----
  relu_bias_bf16<<<dim3((unsigned)(((size_t)Nn * F1 + 255) / 256)), dim3(256), 0, stream>>>(
      h1agg, b1, h1r, (size_t)Nn * F1);
  fill(h2agg, 0.0f, (size_t)Nn * HID);
  fill(m2, -INFINITY, (size_t)Nn);
  fill(d2, 0.0f, (size_t)Nn);

  gemm_wmma<F1, HID><<<dim3(mt, 1), dim3(128), 0, stream>>>(h1r, w2t, h2, Nn);
  att_logits2<<<dim3((Nn + 255) / 256), dim3(256), 0, stream>>>(
      h2, att_src2, att_dst2, as2, ad2, Nn);
  edge_maxk<<<dim3((EE + 255) / 256), dim3(256), 0, stream>>>(ei, E, Nn, 1, as2, ad2, m2);
  edge_expsum<<<dim3((EE + 255) / 256), dim3(256), 0, stream>>>(ei, E, Nn, 1, as2, ad2, m2, d2);
  edge_aggr<<<dim3(EE), dim3(HID), 0, stream>>>(ei, E, Nn, 1, as2, ad2, m2, d2, h2, h2agg);

  // ----- heads -----
  node_scores_k<<<dim3((Nn + 255) / 256), dim3(256), 0, stream>>>(
      h2agg, b2, action_w, action_b, ns, Nn);
  ctx_stop_k<<<dim3(G), dim3(HID), 0, stream>>>(
      h2agg, b2, cur_nodes, cur_counts, stop_w, stop_b, stopsc, Kk);
  softmax_out_k<<<dim3(G), dim3(256), 0, stream>>>(
      ns, masks, stopsc, cur_counts, (float*)d_out, Nn);
}
